// AttentionPooling_65841848648265
// MI455X (gfx1250) — matrix operand order repaired and checked by reading, not compiled, not run on hardware
//
#include <hip/hip_runtime.h>
#include <hip/hip_bf16.h>

static constexpr int kSeg = 4096;
static constexpr int kD   = 256;   // feature dim
static constexpr int kH   = 128;   // hidden dim

typedef __attribute__((ext_vector_type(16))) _Float16 v16h;
typedef __attribute__((ext_vector_type(8)))  float    v8f;

// ---------------------------------------------------------------------------
// Kernel 0: W1 [kD x kH] f32  ->  W1^T [kH x kD] f16 (contiguous K per row)
// ---------------------------------------------------------------------------
__global__ void prep_w1t_kernel(const float* __restrict__ w1,
                                _Float16* __restrict__ w1t) {
  int i = blockIdx.x * 256 + threadIdx.x;       // i in [0, kH*kD)
  if (i >= kH * kD) return;
  int n = i >> 8;                               // hidden col  (0..127)
  int k = i & 255;                              // input row   (0..255)
  w1t[i] = (_Float16)w1[k * kH + n];
}

// ---------------------------------------------------------------------------
// Kernel 1: segment start offsets via binary search (batch is sorted int32)
// seg[s] = lower_bound(batch, s); seg[kSeg] = N
// ---------------------------------------------------------------------------
__global__ void seg_bounds_kernel(const int* __restrict__ batch,
                                  int* __restrict__ seg, int n) {
  int s = blockIdx.x * blockDim.x + threadIdx.x;
  if (s > kSeg) return;
  if (s == kSeg) { seg[s] = n; return; }
  int lo = 0, hi = n;
  while (lo < hi) {
    int mid = (lo + hi) >> 1;
    if (batch[mid] < s) lo = mid + 1; else hi = mid;
  }
  seg[s] = lo;
}

// ---------------------------------------------------------------------------
// Kernel 2: scores[i] = tanh(x[i,:] @ W1 + b1) @ W2 + b2  via f16 WMMA
// One wave per 16-row tile; A fragment reused across all 8 column tiles.
// ---------------------------------------------------------------------------
__global__ __launch_bounds__(256) void score_gemm_kernel(
    const float* __restrict__ x, const _Float16* __restrict__ w1t,
    const float* __restrict__ b1, const float* __restrict__ w2,
    const float* __restrict__ b2, float* __restrict__ scores, int nrows) {
  __shared__ __align__(32) _Float16 sW1T[kH * kD];   // 64 KB

  // cooperative stage of W1^T into LDS (16B vectors)
  {
    const uint4* src = (const uint4*)w1t;
    uint4* dst = (uint4*)sW1T;
    const int nvec = kH * kD * (int)sizeof(_Float16) / 16;  // 4096
    for (int i = threadIdx.x; i < nvec; i += 256) dst[i] = src[i];
  }
  __syncthreads();

  const int wave = threadIdx.x >> 5;
  const int lane = threadIdx.x & 31;
  const int tile = blockIdx.x * 8 + wave;
  const int m0   = tile * 16;
  if (m0 >= nrows) return;                 // uniform per wave -> EXEC all-1s

  const int m  = lane & 15;                // row within tile / col within n-tile
  const int hh = lane >> 4;                // lane half
  const int row = (m0 + m < nrows) ? (m0 + m) : (nrows - 1);
  const float* xrow = x + (size_t)row * kD;

  v8f acc[8] = {};                         // 8 column tiles: full H=128

  #pragma unroll
  for (int c = 0; c < 8; ++c) {            // K loop: 8 chunks of 32
    const int k0 = c * 32;
    // A fragment: lane half hh covers K = 16hh..16hh+15 (contiguous, matches B)
    __align__(16) float tmp[16];
    const float4* p0 = (const float4*)(xrow + k0 + 16 * hh);
    const float4* p1 = (const float4*)(xrow + k0 + 16 * hh + 8);
    *(float4*)(tmp + 0)  = p0[0];
    *(float4*)(tmp + 4)  = p0[1];
    *(float4*)(tmp + 8)  = p1[0];
    *(float4*)(tmp + 12) = p1[1];
    v16h a;
    #pragma unroll
    for (int j = 0; j < 16; ++j) a[j] = (_Float16)tmp[j];

    #pragma unroll
    for (int t = 0; t < 8; ++t) {          // column tiles, reuse A
      const int n = t * 16 + m;            // absolute hidden column
      // B fragment: lane half hh covers K = 16hh..16hh+15, contiguous in W1^T
      const v16h* bp = (const v16h*)(sW1T + n * kD + k0 + 16 * hh);
      acc[t] = __builtin_amdgcn_wmma_f32_16x16x32_f16(
          false, a, false, *bp, (short)0, acc[t], false, false);
    }
  }

  // Epilogue: C layout -> lane owns column (t*16+m) for rows M = r + 8*hh.
  // score[M] = sum_n tanh(h_pre[M][n] + b1[n]) * W2[n] + b2
  float p[8];
  #pragma unroll
  for (int r = 0; r < 8; ++r) p[r] = 0.f;
  #pragma unroll
  for (int t = 0; t < 8; ++t) {
    const int n = t * 16 + m;
    const float b1v = b1[n];
    const float w2v = w2[n];
    #pragma unroll
    for (int r = 0; r < 8; ++r)
      p[r] += tanhf(acc[t][r] + b1v) * w2v;
  }
  // reduce across the 16 lanes of each half (xor 1,2,4,8 stays within half)
  #pragma unroll
  for (int off = 1; off < 16; off <<= 1) {
    #pragma unroll
    for (int r = 0; r < 8; ++r) p[r] += __shfl_xor(p[r], off, 32);
  }
  if (m == 0) {
    const float bb = b2[0];
    #pragma unroll
    for (int r = 0; r < 8; ++r) {
      const int rr = m0 + 8 * hh + r;
      if (rr < nrows) scores[rr] = p[r] + bb;
    }
  }
}

// ---------------------------------------------------------------------------
// Kernel 3: per-segment softmax + weighted pooling.
// One 256-thread block per segment; thread t owns output dimension t.
// pooled[s,d] = (sum_i x[i,d] * exp(score_i - mx)) / (sum_i exp(score_i - mx))
// ---------------------------------------------------------------------------
__global__ __launch_bounds__(256) void pool_kernel(
    const float* __restrict__ x, const float* __restrict__ scores,
    const int* __restrict__ seg, float* __restrict__ out) {
  __shared__ float red[256];
  const int s = blockIdx.x;
  const int t = threadIdx.x;
  const int beg = seg[s], end = seg[s + 1];
  if (beg >= end) { out[(size_t)s * kD + t] = 0.f; return; }  // empty segment

  // segment max
  float mx = -3.402823466e38f;
  for (int i = beg + t; i < end; i += 256) mx = fmaxf(mx, scores[i]);
  red[t] = mx; __syncthreads();
  #pragma unroll
  for (int off = 128; off > 0; off >>= 1) {
    if (t < off) red[t] = fmaxf(red[t], red[t + off]);
    __syncthreads();
  }
  mx = red[0];
  __syncthreads();

  // segment sum of exp
  float sm = 0.f;
  for (int i = beg + t; i < end; i += 256) sm += __expf(scores[i] - mx);
  red[t] = sm; __syncthreads();
  #pragma unroll
  for (int off = 128; off > 0; off >>= 1) {
    if (t < off) red[t] += red[t + off];
    __syncthreads();
  }
  const float inv = 1.0f / red[0];

  // weighted accumulation; x reads fully coalesced across the block
  float acc = 0.f;
  for (int i = beg; i < end; ++i) {
    acc = fmaf(x[(size_t)i * kD + t], __expf(scores[i] - mx), acc);
  }
  out[(size_t)s * kD + t] = acc * inv;
}

// ---------------------------------------------------------------------------
extern "C" void kernel_launch(void* const* d_in, const int* in_sizes, int n_in,
                              void* d_out, int out_size, void* d_ws, size_t ws_size,
                              hipStream_t stream) {
  const float* x     = (const float*)d_in[0];
  const int*   batch = (const int*)d_in[1];
  const float* W1    = (const float*)d_in[2];
  const float* b1    = (const float*)d_in[3];
  const float* W2    = (const float*)d_in[4];
  const float* b2    = (const float*)d_in[5];
  float* out = (float*)d_out;
  const int N = in_sizes[1];               // number of nodes (batch length)

  // workspace layout: scores f32[N] | W1^T f16[kH*kD] | seg int[kSeg+1]
  char* ws = (char*)d_ws;
  float* scores = (float*)ws;
  size_t off = ((size_t)N * sizeof(float) + 255) & ~(size_t)255;
  _Float16* w1t = (_Float16*)(ws + off);
  off += (size_t)kH * kD * sizeof(_Float16);
  off = (off + 255) & ~(size_t)255;
  int* seg = (int*)(ws + off);

  prep_w1t_kernel<<<(kH * kD + 255) / 256, 256, 0, stream>>>(W1, w1t);
  seg_bounds_kernel<<<(kSeg + 1 + 255) / 256, 256, 0, stream>>>(batch, seg, N);

  const int ntiles  = (N + 15) / 16;
  const int nblocks = (ntiles + 7) / 8;    // 8 waves (tiles) per 256-thread block
  score_gemm_kernel<<<nblocks, 256, 0, stream>>>(x, w1t, b1, W2, b2, scores, N);

  pool_kernel<<<kSeg, 256, 0, stream>>>(x, scores, seg, out);
}
